// ReferenceGroupedMLP_87153476371165
// MI455X (gfx1250) — matrix operand order repaired and checked by reading, not compiled, not run on hardware
//
#include <hip/hip_runtime.h>

#define NUM_GROUPS 64
#define IN_DIM 128
#define HID_DIM 256
#define OUT_DIM 128
#define TOKENS_PER_GROUP 8192
#define TILE_B 128   // tokens per workgroup
#define WAVES 8

typedef __attribute__((ext_vector_type(16))) __bf16 v16bf;
typedef __attribute__((ext_vector_type(8)))  float  v8f;

// LDS row strides in bf16 elements; rows are 16B-aligned (multiples of 8)
// so b128 LDS accesses stay aligned; stride mod 64 banks != 0.
#define XS 136
#define GS 136
#define DSTR 72
#define HS 72

#define X0 0
#define G0 (X0 + 128 * XS)              // x tile 128x128 bf16
#define U0 (G0 + 64 * GS)               // gate chunk 64x128 bf16
#define D0 (U0 + 64 * GS)               // up chunk 64x128 bf16
#define H0 (D0 + 128 * DSTR)            // down chunk 128x64 bf16
#define SMEM_U16 (H0 + WAVES * 16 * HS) // 53248 u16 = 106496 B

// bf16 weight workspace layout (u16 elements)
#define WSZ_GATE (NUM_GROUPS * HID_DIM * IN_DIM)   // 2097152
#define WS_BYTES ((size_t)(3 * WSZ_GATE) * 2)      // 12,582,912 B

__device__ __forceinline__ unsigned int pack_bf16x2(float lo, float hi) {
  union { float f; unsigned int u; } a, b;
  a.f = lo; b.f = hi;
  unsigned int ua = a.u + 0x7FFFu + ((a.u >> 16) & 1u);
  unsigned int ub = b.u + 0x7FFFu + ((b.u >> 16) & 1u);
  return (ua >> 16) | (ub & 0xFFFF0000u);
}

__device__ __forceinline__ unsigned short f2bf(float f) {
  union { float f; unsigned int u; } a; a.f = f;
  unsigned int r = a.u + 0x7FFFu + ((a.u >> 16) & 1u);
  return (unsigned short)(r >> 16);
}

union Frag { v16bf v; uint4 q[2]; };

// B fragment (32x16 bf16): two 16B groups with a 16B gap -> 2x ds_load_b128
__device__ __forceinline__ v16bf ldB(const unsigned short* p) {
  Frag f;
  const uint4* q = (const uint4*)p;
  f.q[0] = q[0];
  f.q[1] = q[2];
  return f.v;
}

// A fragment (16x32 bf16): two 16B groups with a 16B gap -> 2x ds_load_b128
__device__ __forceinline__ v16bf ldA(const unsigned short* p) {
  Frag f;
  const uint4* q = (const uint4*)p;
  f.q[0] = q[0];
  f.q[1] = q[2];
  return f.v;
}

// ---------------- prepass: fp32 -> bf16 weight conversion ----------------
__global__ void cvt_bf16(const float* __restrict__ src, unsigned int* __restrict__ dst, int n2) {
  int i = blockIdx.x * blockDim.x + threadIdx.x;
  if (i < n2) {
    float2 v = ((const float2*)src)[i];
    dst[i] = pack_bf16x2(v.x, v.y);
  }
}

// ---------------- main kernel ----------------
template <bool USE_WS>
__global__ __launch_bounds__(256)
void grouped_mlp_wmma(const float* __restrict__ x,
                      const float* __restrict__ gate_w,
                      const float* __restrict__ up_w,
                      const float* __restrict__ down_w,
                      const unsigned short* __restrict__ wbf,
                      float* __restrict__ out) {
  __shared__ __align__(16) unsigned short smem[SMEM_U16];
  unsigned int* s32 = (unsigned int*)smem;

  const int tid  = threadIdx.x;
  const int lane = tid & 31;
  const int wv   = tid >> 5;      // wave id 0..7
  const int hf   = lane >> 4;     // half-wave: 0 or 1
  const int lm   = lane & 15;
  const int g    = blockIdx.y;
  const int tile = blockIdx.x;

  // ---------- stage x tile: 128 tokens x 128 dims, fp32 -> bf16 LDS ----------
  const float4* x4 = (const float4*)x;
  #pragma unroll
  for (int i = 0; i < 16; ++i) {
    int idx = tid + i * 256;            // 0..4095
    int r  = idx >> 5;                  // token row in tile
    int c4 = idx & 31;                  // float4 column
    int b  = tile * TILE_B + r;
    float4 vx = x4[(b * NUM_GROUPS + g) * (IN_DIM / 4) + c4];
    int base = (X0 + r * XS + c4 * 4) >> 1;
    s32[base]     = pack_bf16x2(vx.x, vx.y);
    s32[base + 1] = pack_bf16x2(vx.z, vx.w);
  }
  __syncthreads();

  // ---------- A fragments: this wave's 16 tokens, K = 128 (4 k-steps) ----------
  const unsigned short* ab = smem + X0 + (wv * 16 + lm) * XS + 8 * hf;
  v16bf afr[4];
  #pragma unroll
  for (int ks = 0; ks < 4; ++ks) afr[ks] = ldA(ab + ks * 32);

  const v8f vzero = {0.f, 0.f, 0.f, 0.f, 0.f, 0.f, 0.f, 0.f};
  v8f oacc[8];
  #pragma unroll
  for (int ot = 0; ot < 8; ++ot) oacc[ot] = vzero;

  const float4* g4 = (const float4*)gate_w;
  const float4* u4 = (const float4*)up_w;
  const float4* d4 = (const float4*)down_w;
  const unsigned short* wsg = wbf;
  const unsigned short* wsu = wbf + WSZ_GATE;
  const unsigned short* wsd = wbf + 2 * WSZ_GATE;

  // per-wave fragment base pointers (one address reg each; offsets fold to imm)
  const unsigned short* gb = smem + G0 + lm * GS + 8 * hf;
  const unsigned short* ub = smem + U0 + lm * GS + 8 * hf;
  const unsigned short* db = smem + D0 + lm * DSTR + 8 * hf;
  const unsigned short* hb = smem + H0 + (wv * 16 + lm) * HS + 8 * hf;

  for (int hc = 0; hc < 4; ++hc) {
    __syncthreads();  // previous chunk's compute done before overwriting weights

    if (USE_WS) {
      // ---- async bf16 weight staging: GLOBAL_LOAD_ASYNC_TO_LDS_B128 ----
      const unsigned short* gsrc = wsg + (size_t)(g * HID_DIM + hc * 64) * IN_DIM;
      const unsigned short* usrc = wsu + (size_t)(g * HID_DIM + hc * 64) * IN_DIM;
      const unsigned short* dsrc = wsd + (size_t)g * OUT_DIM * HID_DIM + hc * 64;
      #pragma unroll
      for (int i = 0; i < 4; ++i) {
        int idx = tid + i * 256;        // 0..1023
        int r = idx >> 4, c = idx & 15; // gate/up: row 0..63, 16B unit 0..15
        unsigned lg = 2u * (unsigned)(G0 + r * GS + c * 8);
        unsigned lu = 2u * (unsigned)(U0 + r * GS + c * 8);
        unsigned vo = (unsigned)((r * IN_DIM + c * 8) * 2);
        asm volatile("global_load_async_to_lds_b128 %0, %1, %2"
                     :: "v"(lg), "v"(vo), "s"(gsrc) : "memory");
        asm volatile("global_load_async_to_lds_b128 %0, %1, %2"
                     :: "v"(lu), "v"(vo), "s"(usrc) : "memory");
        int r2 = idx >> 3, c2 = idx & 7; // down: row 0..127, 16B unit 0..7
        unsigned ld = 2u * (unsigned)(D0 + r2 * DSTR + c2 * 8);
        unsigned vd = (unsigned)((r2 * HID_DIM + c2 * 8) * 2);
        asm volatile("global_load_async_to_lds_b128 %0, %1, %2"
                     :: "v"(ld), "v"(vd), "s"(dsrc) : "memory");
      }
      asm volatile("s_wait_asynccnt 0x0" ::: "memory");
    } else {
      // ---- fallback: fp32 weights, convert in VGPRs ----
      #pragma unroll
      for (int i = 0; i < 8; ++i) {
        int idx = tid + i * 256;          // 0..2047
        int r  = idx >> 5;                // hidden row 0..63
        int c4 = idx & 31;                // float4 col over 128 dims
        float4 gv = g4[(g * HID_DIM + hc * 64 + r) * (IN_DIM / 4) + c4];
        float4 uv = u4[(g * HID_DIM + hc * 64 + r) * (IN_DIM / 4) + c4];
        int gbo = (G0 + r * GS + c4 * 4) >> 1;
        s32[gbo]     = pack_bf16x2(gv.x, gv.y);
        s32[gbo + 1] = pack_bf16x2(gv.z, gv.w);
        int ubo = (U0 + r * GS + c4 * 4) >> 1;
        s32[ubo]     = pack_bf16x2(uv.x, uv.y);
        s32[ubo + 1] = pack_bf16x2(uv.z, uv.w);
        int r2  = idx >> 4;               // out row 0..127
        int c42 = idx & 15;               // float4 col over 64 hidden
        float4 dv = d4[(g * OUT_DIM + r2) * (HID_DIM / 4) + hc * 16 + c42];
        int dbo = (D0 + r2 * DSTR + c42 * 4) >> 1;
        s32[dbo]     = pack_bf16x2(dv.x, dv.y);
        s32[dbo + 1] = pack_bf16x2(dv.z, dv.w);
      }
    }
    __syncthreads();

    // ---------- GEMM1: gate & up over this 64-wide hidden chunk ----------
    v8f gacc[4], uacc[4];
    #pragma unroll
    for (int nt = 0; nt < 4; ++nt) { gacc[nt] = vzero; uacc[nt] = vzero; }
    #pragma unroll
    for (int nt = 0; nt < 4; ++nt) {
      #pragma unroll
      for (int ks = 0; ks < 4; ++ks) {
        v16bf bg = ldB(gb + nt * 16 * GS + ks * 32);
        v16bf bu = ldB(ub + nt * 16 * GS + ks * 32);
        gacc[nt] = __builtin_amdgcn_wmma_f32_16x16x32_bf16(
            false, afr[ks], false, bg, (short)0, gacc[nt], false, false);
        uacc[nt] = __builtin_amdgcn_wmma_f32_16x16x32_bf16(
            false, afr[ks], false, bu, (short)0, uacc[nt], false, false);
      }
    }

    // ---------- silu(gate) * up -> bf16 wave-private LDS scratch ----------
    // fast path: g * v_rcp_f32(1 + e^-g) instead of IEEE division
    #pragma unroll
    for (int nt = 0; nt < 4; ++nt) {
      #pragma unroll
      for (int r = 0; r < 8; ++r) {
        float gv = gacc[nt][r];
        float uv = uacc[nt][r];
        float sig = __builtin_amdgcn_rcpf(1.0f + __expf(-gv));
        float hv = gv * sig * uv;
        int m = r + 8 * hf;  // C/D layout: row = r + 8*half, col = lane%16
        smem[H0 + (wv * 16 + m) * HS + nt * 16 + lm] = f2bf(hv);
      }
    }
    // same-wave LDS RAW: compiler inserts s_wait_dscnt

    // ---------- GEMM2: out(16x128) += hidden(16x64) @ down_chunk^T ----------
    #pragma unroll
    for (int ks = 0; ks < 2; ++ks) {
      v16bf a2 = ldA(hb + ks * 32);
      #pragma unroll
      for (int ot = 0; ot < 8; ++ot) {
        v16bf bd = ldB(db + ot * 16 * DSTR + ks * 32);
        oacc[ot] = __builtin_amdgcn_wmma_f32_16x16x32_bf16(
            false, a2, false, bd, (short)0, oacc[ot], false, false);
      }
    }
  }

  // ---------- store fp32 output: 16 tokens x 128 cols per wave ----------
  #pragma unroll
  for (int ot = 0; ot < 8; ++ot) {
    #pragma unroll
    for (int r = 0; r < 8; ++r) {
      int m = r + 8 * hf;
      int b = tile * TILE_B + wv * 16 + m;
      out[(b * NUM_GROUPS + g) * OUT_DIM + ot * 16 + lm] = oacc[ot][r];
    }
  }
}

extern "C" void kernel_launch(void* const* d_in, const int* in_sizes, int n_in,
                              void* d_out, int out_size, void* d_ws, size_t ws_size,
                              hipStream_t stream) {
  const float* x  = (const float*)d_in[0];
  const float* gw = (const float*)d_in[1];
  const float* uw = (const float*)d_in[2];
  const float* dw = (const float*)d_in[3];
  float* out = (float*)d_out;
  dim3 grid(TOKENS_PER_GROUP / TILE_B, NUM_GROUPS);

  const bool use_ws = (d_ws != nullptr) && (ws_size >= WS_BYTES);
  if (use_ws) {
    unsigned int* wsg = (unsigned int*)d_ws;                    // bf16 pairs
    unsigned int* wsu = wsg + WSZ_GATE / 2;
    unsigned int* wsd = wsg + WSZ_GATE;                         // 2*WSZ_GATE u16
    const int n2 = WSZ_GATE / 2;                                // 1,048,576 pairs each
    dim3 cg((n2 + 255) / 256);
    cvt_bf16<<<cg, dim3(256), 0, stream>>>(gw, wsg, n2);
    cvt_bf16<<<cg, dim3(256), 0, stream>>>(uw, wsu, n2);
    cvt_bf16<<<cg, dim3(256), 0, stream>>>(dw, wsd, n2);
    grouped_mlp_wmma<true><<<grid, dim3(256), 0, stream>>>(
        x, gw, uw, dw, (const unsigned short*)d_ws, out);
  } else {
    grouped_mlp_wmma<false><<<grid, dim3(256), 0, stream>>>(
        x, gw, uw, dw, (const unsigned short*)nullptr, out);
  }
}
